// MoE_49675591745736
// MI455X (gfx1250) — matrix operand order repaired and checked by reading, not compiled, not run on hardware
//
#include <hip/hip_runtime.h>

// ---------------- types ----------------
typedef __attribute__((ext_vector_type(16))) __bf16 v16bf;
typedef __attribute__((ext_vector_type(8)))  float  v8f;
typedef __attribute__((ext_vector_type(8)))  __bf16 bf16x8;
typedef __attribute__((ext_vector_type(4)))  __bf16 bf16x4;
typedef __attribute__((ext_vector_type(8)))  float  f32x8;
typedef __attribute__((ext_vector_type(4)))  float  f32x4;

// ---------------- problem constants ----------------
#define B_TOK 16384
#define Dd    256
#define Ff    1024
#define Oo    256
#define NE    8       // total experts (2 shared + 6 specific)
#define NS    2       // shared experts
#define BM    64      // token rows per workgroup
#define FH    512     // F half staged in LDS

__device__ __forceinline__ v8f zero8() {
  v8f z;
#pragma unroll
  for (int i = 0; i < 8; ++i) z[i] = 0.0f;
  return z;
}

// ============================================================
// Kernel 1: gating — softmax over 8 logits, top-2, combine weights
//   combw[b][e] = 1/NS for shared experts, top-k softmax weight for
//   selected routed experts, else 0.  Unifies mean(shared)+weighted(routed)
//   into a single weighted sum over all 8 dense expert outputs.
// ============================================================
__global__ __launch_bounds__(256) void moe_gate_kernel(
    const float* __restrict__ x, const float* __restrict__ gw,
    const float* __restrict__ gb, float* __restrict__ wout,
    float* __restrict__ combw) {
  int b = blockIdx.x * 256 + threadIdx.x;
  if (b >= B_TOK) return;
  float acc[NE];
#pragma unroll
  for (int e = 0; e < NE; ++e) acc[e] = gb[e];
  const float* xr = x + (size_t)b * Dd;
  for (int d = 0; d < Dd; d += 4) {
    f32x4 xv = *(const f32x4*)(xr + d);
#pragma unroll
    for (int j = 0; j < 4; ++j) {
      float xs = xv[j];
#pragma unroll
      for (int e = 0; e < NE; ++e) acc[e] = fmaf(xs, gw[(d + j) * NE + e], acc[e]);
    }
  }
  float m = acc[0];
#pragma unroll
  for (int e = 1; e < NE; ++e) m = fmaxf(m, acc[e]);
  float s = 0.f;
#pragma unroll
  for (int e = 0; e < NE; ++e) { acc[e] = __expf(acc[e] - m); s += acc[e]; }
  float inv = 1.0f / s;
  float w[NE];
#pragma unroll
  for (int e = 0; e < NE; ++e) { w[e] = acc[e] * inv; wout[(size_t)b * NE + e] = w[e]; }
  // top-2 (ties -> lowest index, matching jax.lax.top_k)
  int i1 = 0;
#pragma unroll
  for (int e = 1; e < NE; ++e) if (w[e] > w[i1]) i1 = e;
  int i2 = (i1 == 0) ? 1 : 0;
#pragma unroll
  for (int e = 0; e < NE; ++e) if (e != i1 && w[e] > w[i2]) i2 = e;
  float cw[NE];
#pragma unroll
  for (int e = 0; e < NE; ++e) cw[e] = (e < NS) ? (1.0f / NS) : 0.0f;
  if (i1 >= NS) cw[i1] += w[i1];
  if (i2 >= NS) cw[i2] += w[i2];
#pragma unroll
  for (int e = 0; e < NE; ++e) combw[(size_t)b * NE + e] = cw[e];
}

// ============================================================
// Kernel 2: per-expert (R x C) f32 -> (C x R) bf16 tiled transpose
// ============================================================
__global__ __launch_bounds__(256) void transpose_cvt_kernel(
    const float* __restrict__ srcSh, const float* __restrict__ srcSp,
    __bf16* __restrict__ dst, int R, int C) {
  __shared__ float tile[32][33];
  int e = blockIdx.z;
  const float* src = (e < NS) ? (srcSh + (size_t)e * R * C)
                              : (srcSp + (size_t)(e - NS) * R * C);
  __bf16* d = dst + (size_t)e * R * C;
  int c0 = blockIdx.x * 32, r0 = blockIdx.y * 32;
  int tx = threadIdx.x & 31, ty = threadIdx.x >> 5;  // 32 x 8 threads
#pragma unroll
  for (int i = 0; i < 32; i += 8)
    tile[ty + i][tx] = src[(size_t)(r0 + ty + i) * C + (c0 + tx)];
  __syncthreads();
#pragma unroll
  for (int i = 0; i < 32; i += 8)
    d[(size_t)(c0 + ty + i) * R + (r0 + tx)] = (__bf16)tile[tx][ty + i];
}

// ============================================================
// Kernel 3: fused dense-MoE.
//   Key trick: GEMM1 epilogue scales each h row by the per-token combine
//   weight, so GEMM2 accumulates w_e-weighted expert outputs DIRECTLY into
//   the persistent output accumulators (no per-expert acc, no combine pass).
//   out += (cw·relu(xW1+b1)) @ W2  +  cw[row]·b2[col]
// ============================================================
__global__ __launch_bounds__(256) void moe_main_kernel(
    const float* __restrict__ x,
    const __bf16* __restrict__ w1t,   // [E][F][D] bf16 (transposed)
    const __bf16* __restrict__ w2t,   // [E][O][F] bf16 (transposed)
    const float* __restrict__ sh_b1, const float* __restrict__ sp_b1,
    const float* __restrict__ sh_b2, const float* __restrict__ sp_b2,
    const float* __restrict__ combw,  // [B][E]
    float* __restrict__ out)          // [B][O]
{
  __shared__ __align__(16) __bf16 lds_x[BM * Dd];   // 32 KB
  __shared__ __align__(16) __bf16 lds_h[BM * FH];   // 64 KB
  __shared__ __align__(32) float  lds_cw[NE * BM];  //  2 KB

  const int tid  = threadIdx.x;
  const int wv   = tid >> 5;
  const int lane = tid & 31;
  const int lo16 = lane & 15;
  const int hi   = lane >> 4;
  const int row0 = blockIdx.x * BM;

  // combine-weight tile: lds_cw[e*BM + row]
  for (int i = tid; i < NE * BM; i += 256) {
    int bb = i & (BM - 1);
    int e  = i >> 6;
    lds_cw[e * BM + bb] = combw[(size_t)(row0 + bb) * NE + e];
  }
  // x tile f32 -> bf16 LDS (row-major BM x D, contiguous rows)
  for (int i = tid * 4; i < BM * Dd; i += 1024) {
    f32x4 v = *(const f32x4*)(x + (size_t)row0 * Dd + i);
    bf16x4 hv;
#pragma unroll
    for (int j = 0; j < 4; ++j) hv[j] = (__bf16)v[j];
    *(bf16x4*)(lds_x + i) = hv;
  }
  __syncthreads();

  // persistent weighted-output accumulators: wave covers out cols [wv*32, wv*32+32)
  v8f oacc[2][4];
#pragma unroll
  for (int nt = 0; nt < 2; ++nt)
#pragma unroll
    for (int mt = 0; mt < 4; ++mt) oacc[nt][mt] = zero8();

  for (int e = 0; e < NE; ++e) {
    const __bf16* W1 = w1t + (size_t)e * Ff * Dd;
    const __bf16* W2 = w2t + (size_t)e * Oo * Ff;
    const float*  b1 = (e < NS) ? (sh_b1 + (size_t)e * Ff) : (sp_b1 + (size_t)(e - NS) * Ff);
    const float*  b2 = (e < NS) ? (sh_b2 + (size_t)e * Oo) : (sp_b2 + (size_t)(e - NS) * Oo);

    for (int fh = 0; fh < 2; ++fh) {
      // ---- GEMM1: h[:, fh half] = cw[row] * relu(x @ W1 + b1) -> bf16 LDS ----
      for (int nt = 0; nt < 4; ++nt) {
        const int hn0 = wv * 64 + nt * 16;   // column within half
        const int n0  = fh * FH + hn0;       // column within F
        const float b1v = b1[n0 + lo16];
        v8f a1[4];
#pragma unroll
        for (int mt = 0; mt < 4; ++mt) a1[mt] = zero8();
        for (int k0 = 0; k0 < Dd; k0 += 32) {
          // B fragment: lane holds 16 contiguous K of column n0+lo16
          v16bf bf = *(const v16bf*)(W1 + (size_t)(n0 + lo16) * Dd + k0 + hi * 16);
#pragma unroll
          for (int mt = 0; mt < 4; ++mt) {
            const __bf16* ap = lds_x + (size_t)(mt * 16 + lo16) * Dd + k0 + hi * 16;
            bf16x8 alo = *(const bf16x8*)ap;
            bf16x8 ahi = *(const bf16x8*)(ap + 8);
            v16bf af;
#pragma unroll
            for (int i = 0; i < 8; ++i) { af[i] = alo[i]; af[i + 8] = ahi[i]; }
            a1[mt] = __builtin_amdgcn_wmma_f32_16x16x32_bf16(
                false, af, false, bf, (short)0, a1[mt], false, false);
          }
        }
#pragma unroll
        for (int mt = 0; mt < 4; ++mt) {
          f32x8 cwv = *(const f32x8*)(lds_cw + e * BM + mt * 16 + hi * 8);
#pragma unroll
          for (int r = 0; r < 8; ++r) {
            const int M = mt * 16 + hi * 8 + r;
            float v = fmaxf(a1[mt][r] + b1v, 0.0f) * cwv[r];
            lds_h[(size_t)M * FH + hn0 + lo16] = (__bf16)v;
          }
        }
      }
      __syncthreads();   // h half complete

      // ---- GEMM2: oacc += h_half @ W2[fh half]  (direct accumulation) ----
#pragma unroll
      for (int nt = 0; nt < 2; ++nt) {
        const int n0 = wv * 32 + nt * 16;
        for (int k0 = 0; k0 < FH; k0 += 32) {
          v16bf bf = *(const v16bf*)(W2 + (size_t)(n0 + lo16) * Ff + fh * FH + k0 + hi * 16);
#pragma unroll
          for (int mt = 0; mt < 4; ++mt) {
            const __bf16* ap = lds_h + (size_t)(mt * 16 + lo16) * FH + k0 + hi * 16;
            bf16x8 alo = *(const bf16x8*)ap;
            bf16x8 ahi = *(const bf16x8*)(ap + 8);
            v16bf af;
#pragma unroll
            for (int i = 0; i < 8; ++i) { af[i] = alo[i]; af[i + 8] = ahi[i]; }
            oacc[nt][mt] = __builtin_amdgcn_wmma_f32_16x16x32_bf16(
                false, af, false, bf, (short)0, oacc[nt][mt], false, false);
          }
        }
      }
      __syncthreads();   // done reading h half before next fh/expert overwrites
    }

    // ---- rank-1 bias correction: oacc += cw[row] * b2[col] ----
#pragma unroll
    for (int nt = 0; nt < 2; ++nt) {
      const int n0 = wv * 32 + nt * 16;
      const float b2v = b2[n0 + lo16];
#pragma unroll
      for (int mt = 0; mt < 4; ++mt) {
        f32x8 cwv = *(const f32x8*)(lds_cw + e * BM + mt * 16 + hi * 8);
#pragma unroll
        for (int r = 0; r < 8; ++r)
          oacc[nt][mt][r] = fmaf(cwv[r], b2v, oacc[nt][mt][r]);
      }
    }

    // warm L2 with the next expert's weights (global_prefetch_b8)
    if (e + 1 < NE) {
      const __bf16* nw1 = w1t + (size_t)(e + 1) * Ff * Dd;
      const __bf16* nw2 = w2t + (size_t)(e + 1) * Oo * Ff;
      __builtin_prefetch(nw1 + (size_t)tid * 1024, 0, 1);
      __builtin_prefetch(nw2 + (size_t)tid * 1024, 0, 1);
    }
  }

  // ---- store output tile ----
#pragma unroll
  for (int nt = 0; nt < 2; ++nt)
#pragma unroll
    for (int mt = 0; mt < 4; ++mt)
#pragma unroll
      for (int r = 0; r < 8; ++r) {
        const int M = mt * 16 + hi * 8 + r;
        out[(size_t)(row0 + M) * Oo + wv * 32 + nt * 16 + lo16] = oacc[nt][mt][r];
      }
}

// ============================================================
extern "C" void kernel_launch(void* const* d_in, const int* in_sizes, int n_in,
                              void* d_out, int out_size, void* d_ws, size_t ws_size,
                              hipStream_t stream) {
  const float* x     = (const float*)d_in[0];
  const float* gw    = (const float*)d_in[1];
  const float* gb    = (const float*)d_in[2];
  const float* sh_w1 = (const float*)d_in[3];
  const float* sh_b1 = (const float*)d_in[4];
  const float* sh_w2 = (const float*)d_in[5];
  const float* sh_b2 = (const float*)d_in[6];
  const float* sp_w1 = (const float*)d_in[7];
  const float* sp_b1 = (const float*)d_in[8];
  const float* sp_w2 = (const float*)d_in[9];
  const float* sp_b2 = (const float*)d_in[10];

  float* out0 = (float*)d_out;                       // (B, O)
  float* out1 = out0 + (size_t)B_TOK * Oo;           // (B, E) softmax weights

  char*   ws    = (char*)d_ws;
  float*  combw = (float*)ws;            ws += (size_t)B_TOK * NE * sizeof(float);
  __bf16* w1t   = (__bf16*)ws;           ws += (size_t)NE * Ff * Dd * sizeof(__bf16);
  __bf16* w2t   = (__bf16*)ws;           ws += (size_t)NE * Oo * Ff * sizeof(__bf16);
  (void)ws_size; (void)in_sizes; (void)n_in; (void)out_size;

  moe_gate_kernel<<<B_TOK / 256, 256, 0, stream>>>(x, gw, gb, out1, combw);
  // W1: (D x F) -> bf16 (F x D)   grid (C/32, R/32, E)
  transpose_cvt_kernel<<<dim3(Ff / 32, Dd / 32, NE), 256, 0, stream>>>(sh_w1, sp_w1, w1t, Dd, Ff);
  // W2: (F x O) -> bf16 (O x F)
  transpose_cvt_kernel<<<dim3(Oo / 32, Ff / 32, NE), 256, 0, stream>>>(sh_w2, sp_w2, w2t, Ff, Oo);
  moe_main_kernel<<<B_TOK / BM, 256, 0, stream>>>(x, w1t, w2t, sh_b1, sp_b1,
                                                  sh_b2, sp_b2, combw, out0);
}
